// MultiHeadAttention_19250043421085
// MI455X (gfx1250) — matrix operand order repaired and checked by reading, not compiled, not run on hardware
//
#include <hip/hip_runtime.h>
#include <hip/hip_bf16.h>

// Problem constants (from reference)
#define NH     16       // heads
#define DMODEL 1024
#define DKK    64       // d_k
#define SEQ    1024
#define BSZ    64
#define MTOT   (SEQ * BSZ)   // 65536 rows for the big GEMMs

typedef __attribute__((ext_vector_type(16))) __bf16 v16bf;
typedef __attribute__((ext_vector_type(8)))  float  v8f;

union FragU {
    v16bf  v;
    uint4  q[2];
    __bf16 e[16];
};

// A-matrix fragment, 16x32 bf16 (ISA 7.12.2): lane<16 holds row=lane,
// K = {0..7, 16..23}; lane>=16 holds same row, K = {8..15, 24..31}.
__device__ __forceinline__ v16bf load_a_frag(const __bf16* base, int ld) {
    const int lane = threadIdx.x & 31;
    const int row  = lane & 15;
    const int k0   = (lane >> 4) << 3;     // 0 or 8
    FragU f;
    f.q[0] = *(const uint4*)(base + row * ld + k0);
    f.q[1] = *(const uint4*)(base + row * ld + k0 + 16);
    return f.v;
}

// B-matrix fragment, 32x16 bf16: lane<16 holds column n=lane
// with K = {0..7, 16..23}; lane>=16 holds n=lane-16 with K = {8..15, 24..31}.
// Source is [n][k] row-major (= W[e][d] natural layout).
__device__ __forceinline__ v16bf load_b_frag(const __bf16* base, int ld) {
    const int lane = threadIdx.x & 31;
    const int row  = lane & 15;
    const int k0   = (lane >> 4) << 3;     // 0 or 8
    FragU f;
    f.q[0] = *(const uint4*)(base + row * ld + k0);
    f.q[1] = *(const uint4*)(base + row * ld + k0 + 16);
    return f.v;
}

__device__ __forceinline__ v8f wmma_bf(v16bf a, v16bf b, v8f c) {
    return __builtin_amdgcn_wmma_f32_16x16x32_bf16(
        false, a, false, b, (short)0, c, false, false);
}

__device__ __forceinline__ void cvt16(const float4* f, FragU& t) {
#pragma unroll
    for (int i = 0; i < 4; ++i) {
        t.e[4 * i + 0] = (__bf16)f[i].x;
        t.e[4 * i + 1] = (__bf16)f[i].y;
        t.e[4 * i + 2] = (__bf16)f[i].z;
        t.e[4 * i + 3] = (__bf16)f[i].w;
    }
}

// ---- CDNA5 async global->LDS copy (GLOBAL_LOAD_ASYNC_TO_LDS_B128, ASYNCcnt)
// AS(3) pointers are LDS byte offsets (the VDST VGPR supplies the LDS address).
__device__ __forceinline__ unsigned lds_off_of(void* p) {
    return (unsigned)(uintptr_t)(__attribute__((address_space(3))) void*)p;
}
__device__ __forceinline__ void async_ld_b128(void* lds_dst, const void* gsrc) {
    asm volatile("global_load_async_to_lds_b128 %0, %1, off"
                 :: "v"(lds_off_of(lds_dst)),
                    "v"((unsigned long long)(uintptr_t)gsrc)
                 : "memory");
}
__device__ __forceinline__ void wait_async0() {
    asm volatile("s_wait_asynccnt 0x0" ::: "memory");
}

// ---------------------------------------------------------------- convert
__global__ __launch_bounds__(256) void cvt_f32_bf16(
    const float* __restrict__ in, __bf16* __restrict__ out, int n) {
    int i = (blockIdx.x * 256 + threadIdx.x) * 4;
    if (i + 3 < n) {
        float4 v = *(const float4*)(in + i);
        __bf16 o[4] = {(__bf16)v.x, (__bf16)v.y, (__bf16)v.z, (__bf16)v.w};
        *(uint2*)(out + i) = *(const uint2*)o;
    } else {
        for (; i < n; ++i) out[i] = (__bf16)in[i];
    }
}

// ---------------------------------------------------------------- GEMM
// C[M,N] = A[M,K] * B^T + bias, B given as [n][k] row-major (= W[e][d]),
// K = N = 1024, M = 65536.  A is f32 (converted during staging) or bf16
// (staged via async DMA).  Double-buffered LDS pipeline, one barrier/K-step.
// OUT_MODE 0: f32 natural [m][n]
// OUT_MODE 1: bf16 [s][h][b][d]  (m = s*64+b, n = h*64+d)   -> Q, K
// OUT_MODE 2: bf16 [s][h][d][b]  (transposed per head)       -> V
template <bool A_F32, int OUT_MODE>
__global__ __launch_bounds__(256) void gemm_wmma(
    const void* __restrict__ Aptr, const __bf16* __restrict__ B,
    const float* __restrict__ bias, void* __restrict__ Out) {
    constexpr int K = DMODEL, N = DMODEL;
    constexpr int BM = 128, BN = 128, BK = 32;
    constexpr int LDA = BK + 8;             // LDS pad vs bank conflicts
    constexpr int NK  = K / BK;             // 32
    constexpr int STG = BM * LDA;           // 5120 bf16 per stage buffer
    constexpr int LDT = 136;                // epilogue repack stride

    // 40 KB: 4 stage buffers; aliased by the 128x136 bf16 epilogue tile
    __shared__ __align__(16) __bf16 smem[4 * STG];

    const int tid   = threadIdx.x;
    const int wave  = tid >> 5;
    const int wm    = wave & 3;             // 4 waves along M
    const int wn    = wave >> 2;            // 2 waves along N
    const int m_blk = blockIdx.y * BM;
    const int n_blk = blockIdx.x * BN;      // x fastest: A band reuse in L2

    const int srow = tid >> 1;              // staging: row 0..127
    const int scol = (tid & 1) * 16;        // staging: col 0 or 16

    const float*  Af = (const float*)Aptr;
    const __bf16* Ab = (const __bf16*)Aptr;

    v8f acc[2][4] = {};
    float4 fa[4];                           // prefetch regs (A as f32 only)

    // ---- stage k-step 0 into buffer 0
    {
        __bf16* dA = smem;
        __bf16* dB = smem + 2 * STG;
        if (A_F32) {
            const float* src = Af + (size_t)(m_blk + srow) * K + scol;
#pragma unroll
            for (int i = 0; i < 4; ++i) fa[i] = *(const float4*)(src + 4 * i);
            FragU t; cvt16(fa, t);
            *(uint4*)&dA[srow * LDA + scol]     = t.q[0];
            *(uint4*)&dA[srow * LDA + scol + 8] = t.q[1];
        } else {
            const __bf16* src = Ab + (size_t)(m_blk + srow) * K + scol;
            async_ld_b128(&dA[srow * LDA + scol],     src);
            async_ld_b128(&dA[srow * LDA + scol + 8], src + 8);
        }
        const __bf16* sb = B + (size_t)(n_blk + srow) * K + scol;
        async_ld_b128(&dB[srow * LDA + scol],     sb);
        async_ld_b128(&dB[srow * LDA + scol + 8], sb + 8);
        wait_async0();
    }
    __syncthreads();

    for (int kt = 0; kt < NK; ++kt) {
        __bf16* cA = smem + (kt & 1) * STG;
        __bf16* cB = smem + 2 * STG + (kt & 1) * STG;
        __bf16* nA = smem + ((kt + 1) & 1) * STG;
        __bf16* nB = smem + 2 * STG + ((kt + 1) & 1) * STG;

        // ---- issue copies for k-step kt+1 (overlap with WMMAs below).
        // Readers of the target buffer finished before the barrier that
        // ended iteration kt-1, so async writes into it are safe now.
        if (kt + 1 < NK) {
            const int kk = (kt + 1) * BK;
            if (A_F32) {
                const float* src = Af + (size_t)(m_blk + srow) * K + kk + scol;
#pragma unroll
                for (int i = 0; i < 4; ++i) fa[i] = *(const float4*)(src + 4 * i);
            } else {
                const __bf16* src = Ab + (size_t)(m_blk + srow) * K + kk + scol;
                async_ld_b128(&nA[srow * LDA + scol],     src);
                async_ld_b128(&nA[srow * LDA + scol + 8], src + 8);
            }
            const __bf16* sb = B + (size_t)(n_blk + srow) * K + kk + scol;
            async_ld_b128(&nB[srow * LDA + scol],     sb);
            async_ld_b128(&nB[srow * LDA + scol + 8], sb + 8);
        }

        // ---- 8 WMMAs from current LDS buffers
        v16bf a0 = load_a_frag(&cA[(wm * 32) * LDA], LDA);
        v16bf a1 = load_a_frag(&cA[(wm * 32 + 16) * LDA], LDA);
        v16bf b0 = load_b_frag(&cB[(wn * 64) * LDA], LDA);
        v16bf b1 = load_b_frag(&cB[(wn * 64 + 16) * LDA], LDA);
        v16bf b2 = load_b_frag(&cB[(wn * 64 + 32) * LDA], LDA);
        v16bf b3 = load_b_frag(&cB[(wn * 64 + 48) * LDA], LDA);

        acc[0][0] = wmma_bf(a0, b0, acc[0][0]);
        acc[0][1] = wmma_bf(a0, b1, acc[0][1]);
        acc[0][2] = wmma_bf(a0, b2, acc[0][2]);
        acc[0][3] = wmma_bf(a0, b3, acc[0][3]);
        acc[1][0] = wmma_bf(a1, b0, acc[1][0]);
        acc[1][1] = wmma_bf(a1, b1, acc[1][1]);
        acc[1][2] = wmma_bf(a1, b2, acc[1][2]);
        acc[1][3] = wmma_bf(a1, b3, acc[1][3]);

        // ---- finish staging of the next buffer, then publish it
        if (kt + 1 < NK) {
            if (A_F32) {
                FragU t; cvt16(fa, t);
                *(uint4*)&nA[srow * LDA + scol]     = t.q[0];
                *(uint4*)&nA[srow * LDA + scol + 8] = t.q[1];
            }
            wait_async0();
        }
        __syncthreads();
    }

    // ---- epilogue.  C 16x16 f32 layout: VGPR r, lanes 0-15 -> (M=r, N=lane),
    // lanes 16-31 -> (M=r+8, N=lane-16).
    const int lane = tid & 31;
    const int rofs = (lane >> 4) << 3;
    const int cofs = lane & 15;

    if (OUT_MODE == 0) {
        // f32 natural: half-wave rows are 64B contiguous; lines filled by j-loop
#pragma unroll
        for (int i = 0; i < 2; ++i) {
#pragma unroll
            for (int j = 0; j < 4; ++j) {
#pragma unroll
                for (int r = 0; r < 8; ++r) {
                    const int row = m_blk + wm * 32 + i * 16 + r + rofs;
                    const int col = n_blk + wn * 64 + j * 16 + cofs;
                    ((float*)Out)[(size_t)row * N + col] = acc[i][j][r] + bias[col];
                }
            }
        }
    } else {
        // repack through LDS (aliases the stage buffers; last loop iteration
        // ended with a barrier), then emit 128-bit coalesced stores.
        __bf16* sT = smem;
#pragma unroll
        for (int i = 0; i < 2; ++i) {
#pragma unroll
            for (int j = 0; j < 4; ++j) {
#pragma unroll
                for (int r = 0; r < 8; ++r) {
                    const int row = wm * 32 + i * 16 + r + rofs;   // local m
                    const int col = wn * 64 + j * 16 + cofs;       // local n
                    const float v = acc[i][j][r] + bias[n_blk + col];
                    if (OUT_MODE == 1) sT[row * LDT + col] = (__bf16)v;
                    else               sT[col * LDT + row] = (__bf16)v;
                }
            }
        }
        __syncthreads();

        const int s0 = m_blk >> 6;          // 2 s-values per block
        const int h0 = n_blk >> 6;          // 2 h-values per block
#pragma unroll
        for (int it = 0; it < 8; ++it) {
            const int cid  = it * 256 + tid;    // 2048 16-byte chunks
            const int line = cid >> 4;          // 0..127
            const int part = cid & 15;
            const int grp  = part >> 3;         // which 64-group
            const int e8   = (part & 7) * 8;    // element offset in group
            const uint4 val = *(const uint4*)&sT[line * LDT + grp * 64 + e8];
            if (OUT_MODE == 1) {
                // line = local m: s = s0+(line>>6), b = line&63; grp -> h; e8 -> d
                const int s = s0 + (line >> 6), b = line & 63, h = h0 + grp;
                __bf16* dst = (__bf16*)Out +
                    ((((size_t)s * NH + h) * BSZ + b) * DKK + e8);
                *(uint4*)dst = val;
            } else {
                // line = local n: h = h0+(line>>6), d = line&63; grp -> s; e8 -> b
                const int h = h0 + (line >> 6), d = line & 63, s = s0 + grp;
                __bf16* dst = (__bf16*)Out +
                    ((((size_t)s * NH + h) * DKK + d) * BSZ + e8);
                *(uint4*)dst = val;
            }
        }
    }
}

// ---------------------------------------------------------------- attention
// One workgroup (4 waves) per (s,h).  scores = Q K^T / 8, mask, attn = S*V,
// softmax over d_k, store bf16 into concat layout [s][b][h*64+d].
__global__ __launch_bounds__(128) void attn_kernel(
    const __bf16* __restrict__ Qp, const __bf16* __restrict__ Kp,
    const __bf16* __restrict__ Vt, const int* __restrict__ mask,
    __bf16* __restrict__ concat) {
    const int s    = blockIdx.x >> 4;
    const int h    = blockIdx.x & 15;
    const int tid  = threadIdx.x;
    const int lane = tid & 31;
    const int wave = tid >> 5;                  // 0..3, 16 M-rows each

    __shared__ int msum;
    constexpr int LDS_S = 72;                   // bf16 scores stride
    constexpr int LDS_F = 68;                   // f32 attn stride
    // sS (9.2 KB) and sF (17.4 KB) are live in disjoint phases -> alias them
    __shared__ __align__(16) char asmem[64 * LDS_F * 4];
    __bf16* sS = (__bf16*)asmem;
    float*  sF = (float*)asmem;

    if (tid == 0) msum = 0;
    __syncthreads();
    if (tid < 64) {
        const int mv = mask[tid * SEQ + s];     // src_mask[b][s]
        atomicAdd(&msum, mv * mv);
    }

    const size_t hb = ((size_t)s * NH + h) * (size_t)(BSZ * DKK);
    const __bf16* Q  = Qp + hb;                 // [b][d]
    const __bf16* Kh = Kp + hb;                 // [c][d] == B [n][k]
    const __bf16* Vh = Vt + hb;                 // [d][c] == B [n][k]

    // scores = Q * K^T  (M=64 N=64 K=64), wave handles rows [16*wave, +16)
    v8f acc[4] = {};
#pragma unroll
    for (int ks = 0; ks < 2; ++ks) {
        const int k0 = ks * 32;
        v16bf a = load_a_frag(Q + (size_t)(wave * 16) * DKK + k0, DKK);
#pragma unroll
        for (int j = 0; j < 4; ++j) {
            v16bf b = load_b_frag(Kh + (size_t)(j * 16) * DKK + k0, DKK);
            acc[j] = wmma_bf(a, b, acc[j]);
        }
    }
    __syncthreads();                            // msum complete
    const bool masked = (msum == 0);

    const int rofs = (lane >> 4) << 3;
    const int cofs = lane & 15;
#pragma unroll
    for (int j = 0; j < 4; ++j) {
#pragma unroll
        for (int r = 0; r < 8; ++r) {
            const int row = wave * 16 + r + rofs;
            const int col = j * 16 + cofs;
            const float v = masked ? -1e9f : acc[j][r] * 0.125f;
            sS[row * LDS_S + col] = (__bf16)v;
        }
    }
    __syncthreads();

    // attn = scores * V   (V supplied transposed: [n=d][k=c])
    v8f acc2[4] = {};
#pragma unroll
    for (int ks = 0; ks < 2; ++ks) {
        const int k0 = ks * 32;
        v16bf a = load_a_frag(&sS[(wave * 16) * LDS_S + k0], LDS_S);
#pragma unroll
        for (int j = 0; j < 4; ++j) {
            v16bf b = load_b_frag(Vh + (size_t)(j * 16) * DKK + k0, DKK);
            acc2[j] = wmma_bf(a, b, acc2[j]);
        }
    }
    __syncthreads();                            // done reading sS -> reuse as sF
#pragma unroll
    for (int j = 0; j < 4; ++j)
#pragma unroll
        for (int r = 0; r < 8; ++r)
            sF[(wave * 16 + r + rofs) * LDS_F + (j * 16 + cofs)] = acc2[j][r];
    __syncthreads();

    // softmax over d_k (64) per row b; vectorized bf16 stores to concat
    if (tid < 64) {
        const int b = tid;
        float ev[64];
        float mx = -3.4e38f;
#pragma unroll
        for (int d4 = 0; d4 < 64; d4 += 4) {
            const float4 v = *(const float4*)&sF[b * LDS_F + d4];
            ev[d4 + 0] = v.x; ev[d4 + 1] = v.y;
            ev[d4 + 2] = v.z; ev[d4 + 3] = v.w;
            mx = fmaxf(mx, fmaxf(fmaxf(v.x, v.y), fmaxf(v.z, v.w)));
        }
        float sum = 0.f;
#pragma unroll
        for (int d = 0; d < 64; ++d) {
            ev[d] = __expf(ev[d] - mx);
            sum += ev[d];
        }
        const float inv = __frcp_rn(sum);
        __bf16* dst = concat + ((size_t)s * BSZ + b) * DMODEL + h * DKK;
#pragma unroll
        for (int d8 = 0; d8 < 64; d8 += 8) {
            __bf16 o[8];
#pragma unroll
            for (int u = 0; u < 8; ++u) o[u] = (__bf16)(ev[d8 + u] * inv);
            *(uint4*)(dst + d8) = *(const uint4*)o;
        }
    }
}

// ---------------------------------------------------------------- launch
extern "C" void kernel_launch(void* const* d_in, const int* in_sizes, int n_in,
                              void* d_out, int out_size, void* d_ws, size_t ws_size,
                              hipStream_t stream) {
    const float* k_in = (const float*)d_in[0];
    const float* v_in = (const float*)d_in[1];
    const int*   mask = (const int*)d_in[2];
    const float* gt   = (const float*)d_in[3];
    const float* Wq   = (const float*)d_in[4];
    const float* bq   = (const float*)d_in[5];
    const float* Wk   = (const float*)d_in[6];
    const float* bk   = (const float*)d_in[7];
    const float* Wv   = (const float*)d_in[8];
    const float* bv   = (const float*)d_in[9];
    const float* Wo   = (const float*)d_in[10];
    const float* bo   = (const float*)d_in[11];

    // workspace layout (bf16 elements)
    const size_t NW   = (size_t)DMODEL * DMODEL;          // 1M per weight
    const size_t NBIG = (size_t)SEQ * NH * BSZ * DKK;     // 64M per tensor
    __bf16* ws     = (__bf16*)d_ws;
    __bf16* wqb    = ws;
    __bf16* wkb    = wqb + NW;
    __bf16* wvb    = wkb + NW;
    __bf16* wob    = wvb + NW;
    __bf16* qp     = wob + NW;          // [s][h][b][d]
    __bf16* kp     = qp + NBIG;         // [s][h][c][d]
    __bf16* vtp    = kp + NBIG;         // [s][h][d][c]
    __bf16* concat = vtp + NBIG;        // [s][b][h*64+d]

    // 1) weights -> bf16 (natural [e][d] layout == B operand [n][k])
    {
        const int n = (int)NW;
        const int blocks = (n / 4 + 255) / 256;
        cvt_f32_bf16<<<blocks, 256, 0, stream>>>(Wq, wqb, n);
        cvt_f32_bf16<<<blocks, 256, 0, stream>>>(Wk, wkb, n);
        cvt_f32_bf16<<<blocks, 256, 0, stream>>>(Wv, wvb, n);
        cvt_f32_bf16<<<blocks, 256, 0, stream>>>(Wo, wob, n);
    }

    // 2) projections (A is f32, converted during LDS staging; B async-DMA'd)
    dim3 grid(DMODEL / 128, MTOT / 128);    // x fastest -> A band reuse in L2
    gemm_wmma<true, 1><<<grid, 256, 0, stream>>>(gt,   wqb, bq, qp);
    gemm_wmma<true, 1><<<grid, 256, 0, stream>>>(k_in, wkb, bk, kp);
    gemm_wmma<true, 2><<<grid, 256, 0, stream>>>(v_in, wvb, bv, vtp);

    // 3) per-(s,h) attention (scores, mask, scores*V, softmax) -> concat
    attn_kernel<<<SEQ * NH, 128, 0, stream>>>(qp, kp, vtp, mask, concat);

    // 4) output projection (A bf16 async-DMA'd), f32 natural into d_out
    gemm_wmma<false, 0><<<grid, 256, 0, stream>>>(concat, wob, bo, d_out);
}
